// Model_56238301774256
// MI455X (gfx1250) — compile-verified, matrix operand order checked by reading, not run on hardware
//
#include <hip/hip_runtime.h>
#include <hip/hip_bf16.h>

typedef __attribute__((ext_vector_type(16))) _Float16 v16h;
typedef __attribute__((ext_vector_type(8)))  float    v8f;

constexpr int B_SZ = 8, L = 2048, D_MODEL = 512, D_STATE = 16;
constexpr int D_INNER = 1024, DT_RANK = 32;
constexpr int ENC_IN = 7, MARK_IN = 4, C_OUT = 7, PRED_LEN = 96;
constexpr int BL = B_SZ * L;            // 16384 rows
constexpr int CHUNK = 128, NCH = L / CHUNK, NCH1 = NCH - 1;   // 16 chunks; pass1 does 15

// ---------------------------------------------------------------------------
// WMMA GEMM: C(MxN) = A(f32,MxK) * B(f32,KxN), f16 multiply / f32 accumulate.
// Block = 8 waves; block computes 128 rows x 16 cols. The 16-col B chunk is
// staged once per K-step into LDS as f16 (double-buffered, 1 barrier/chunk)
// and shared by all 8 waves; each wave owns a distinct 16-row A strip.
// Fragment layouts per CDNA5 ISA 7.12.2 (wave32):
//   A: row m = lane&15; e<8 -> K = kh*8+e, e>=8 -> K = 16+kh*8+(e-8)
//   B: col n = lane&15; same K striping (transpose-symmetric)
//   C/D: col n = lane&15; VGPR r -> row r + 8*kh
// ---------------------------------------------------------------------------
__global__ void k_gemm(const float* __restrict__ A, const float* __restrict__ B,
                       float* __restrict__ C, int lda, int ldb, int ldc, int K) {
  __shared__ _Float16 lb[2][16 * 40];           // [buf][n*40 + k], padded stride
  const int wave = threadIdx.x >> 5;
  const int lane = threadIdx.x & 31;
  const int mr   = lane & 15;
  const int kh   = (lane >> 4) & 1;
  const int M0   = (blockIdx.x * 8 + wave) * 16;
  const int N0   = blockIdx.y * 16;
  const float* arow = A + (size_t)(M0 + mr) * lda;
  const int sn = threadIdx.x & 15;              // staging: col
  const int sk = threadIdx.x >> 4;              // staging: k (0..15)
  v8f c = {0.f, 0.f, 0.f, 0.f, 0.f, 0.f, 0.f, 0.f};
#pragma unroll 2
  for (int kb = 0; kb < K; kb += 32) {
    const int buf = (kb >> 5) & 1;
    lb[buf][sn * 40 + sk]      = (_Float16)B[(size_t)(kb + sk) * ldb + N0 + sn];
    lb[buf][sn * 40 + 16 + sk] = (_Float16)B[(size_t)(kb + 16 + sk) * ldb + N0 + sn];
    __syncthreads();
    if (kb + 32 < K) __builtin_prefetch(arow + kb + 32, 0, 0);
    v16h a, b;
#pragma unroll
    for (int e = 0; e < 8; ++e) {
      a[e]     = (_Float16)arow[kb + kh * 8 + e];
      a[8 + e] = (_Float16)arow[kb + 16 + kh * 8 + e];
    }
    const _Float16* col = &lb[buf][mr * 40];
#pragma unroll
    for (int e = 0; e < 8; ++e) {
      b[e]     = col[kh * 8 + e];
      b[8 + e] = col[16 + kh * 8 + e];
    }
    c = __builtin_amdgcn_wmma_f32_16x16x32_f16(false, a, false, b,
                                               (short)0, c, false, false);
  }
#pragma unroll
  for (int r = 0; r < 8; ++r)
    C[(size_t)(M0 + r + 8 * kh) * ldc + (N0 + mr)] = c[r];
}

// ---------------------------------------------------------------------------
// X = concat(x_enc, x_mark_enc) @ W_in + b_in        (BL x 512)
// ---------------------------------------------------------------------------
__global__ void k_in(const float* __restrict__ xe, const float* __restrict__ xm,
                     const float* __restrict__ Win, const float* __restrict__ bin,
                     float* __restrict__ X) {
  const size_t bl = blockIdx.x;
  const int d = blockIdx.y * 256 + threadIdx.x;
  float acc = bin[d];
#pragma unroll
  for (int k = 0; k < ENC_IN; ++k)  acc += xe[bl * ENC_IN + k]  * Win[k * D_MODEL + d];
#pragma unroll
  for (int k = 0; k < MARK_IN; ++k) acc += xm[bl * MARK_IN + k] * Win[(ENC_IN + k) * D_MODEL + d];
  X[bl * D_MODEL + d] = acc;
}

// ---------------------------------------------------------------------------
// Depthwise causal conv (D_CONV=4) + bias + SiLU on u = XZ[:, :, :1024]
// ---------------------------------------------------------------------------
__global__ void k_conv(const float* __restrict__ XZ, const float* __restrict__ cw,
                       const float* __restrict__ cb, float* __restrict__ UC) {
  const int blx = blockIdx.x;                 // b*L + l
  const int b = blx >> 11, l = blx & (L - 1);
  const int d = blockIdx.y * 256 + threadIdx.x;
  float acc = cb[d];
#pragma unroll
  for (int j = 0; j < 4; ++j) {
    const int ll = l - 3 + j;
    if (ll >= 0)
      acc += XZ[((size_t)b * L + ll) * (2 * D_INNER) + d] * cw[d * 4 + j];
  }
  UC[(size_t)blx * D_INNER + d] = acc / (1.f + expf(-acc));   // SiLU
}

// ---------------------------------------------------------------------------
// XD = UC @ W_xproj   (BL x 64) : [dt_lo(32) | B(16) | C(16)]
// ---------------------------------------------------------------------------
__global__ void k_xproj(const float* __restrict__ UC, const float* __restrict__ Wx,
                        float* __restrict__ XD) {
  __shared__ float row[D_INNER];
  const size_t bl = blockIdx.x;
  for (int k = threadIdx.x; k < D_INNER; k += 64) row[k] = UC[bl * D_INNER + k];
  __syncthreads();
  const int n = threadIdx.x;
  float acc = 0.f;
  for (int k = 0; k < D_INNER; ++k) acc += row[k] * Wx[k * 64 + n];
  XD[bl * 64 + n] = acc;
}

// ---------------------------------------------------------------------------
// DT = softplus(dt_lo @ W_dt + b_dt)   (BL x 1024)
// ---------------------------------------------------------------------------
__global__ void k_dt(const float* __restrict__ XD, const float* __restrict__ Wdt,
                     const float* __restrict__ bdt, float* __restrict__ DTo) {
  __shared__ float r[DT_RANK];
  const size_t bl = blockIdx.x;
  const int d = blockIdx.y * 256 + threadIdx.x;
  if (threadIdx.x < DT_RANK) r[threadIdx.x] = XD[bl * 64 + threadIdx.x];
  __syncthreads();
  float acc = bdt[d];
#pragma unroll
  for (int k = 0; k < DT_RANK; ++k) acc += r[k] * Wdt[k * D_INNER + d];
  DTo[bl * D_INNER + d] = (acc > 20.f) ? acc : log1pf(expf(acc));
}

// ---------------------------------------------------------------------------
// Chunked selective scan, pass 1: chunks 0..14 run in parallel with h0 = 0.
// Records per-(channel,chunk) final state h_f[16] and sum(dt) (the chunk's
// state-transition is exp(A * sum(dt)) by linearity). No y needed -> only B
// staged in LDS. grid = (B*4, 15), block = 256 (one lane per channel).
// ---------------------------------------------------------------------------
__global__ void k_scan1(const float* __restrict__ UC, const float* __restrict__ DT,
                        const float* __restrict__ XD, const float* __restrict__ A_log,
                        float* __restrict__ HF, float* __restrict__ SDT) {
  __shared__ float sB[2][16];
  const int bx = blockIdx.x;
  const int b = bx >> 2;
  const int d = (bx & 3) * 256 + threadIdx.x;
  const int c = blockIdx.y;                    // chunk 0..14
  float Aa[D_STATE], h[D_STATE];
#pragma unroll
  for (int s = 0; s < D_STATE; ++s) {
    Aa[s] = -expf(A_log[d * D_STATE + s]);
    h[s] = 0.f;
  }
  float sdt = 0.f;
  const int l0 = c * CHUNK;
  for (int i = 0; i < CHUNK; ++i) {
    const size_t bl = (size_t)b * L + (l0 + i);
    const int buf = i & 1;
    if (threadIdx.x < 16) sB[buf][threadIdx.x] = XD[bl * 64 + DT_RANK + threadIdx.x];
    if (threadIdx.x == 16 && i + 1 < CHUNK)
      __builtin_prefetch(XD + (bl + 1) * 64 + DT_RANK, 0, 0);
    __syncthreads();
    const float dt = DT[bl * D_INNER + d];
    const float u  = UC[bl * D_INNER + d];
    sdt += dt;
#pragma unroll
    for (int s = 0; s < D_STATE; ++s)
      h[s] = __expf(dt * Aa[s]) * h[s] + dt * sB[buf][s] * u;
  }
  const size_t idx = (size_t)(b * NCH1 + c) * D_INNER + d;
  SDT[idx] = sdt;
#pragma unroll
  for (int s = 0; s < D_STATE; ++s) HF[idx * D_STATE + s] = h[s];
}

// ---------------------------------------------------------------------------
// Pass 2: fold the 15 chunk summaries (h = exp(A*sdt_c)*h + hf_c) to obtain
// h at l = 1920, then replay chunk 15 (contains the last 96 steps) emitting
// y * silu(z). grid = B*4, block = 256.
// ---------------------------------------------------------------------------
__global__ void k_scan2(const float* __restrict__ XZ, const float* __restrict__ UC,
                        const float* __restrict__ DT, const float* __restrict__ XD,
                        const float* __restrict__ A_log, const float* __restrict__ Dvec,
                        const float* __restrict__ HF, const float* __restrict__ SDT,
                        float* __restrict__ YL) {
  __shared__ float sbc[2][32];                 // [buf][ B(16) | C(16) ]
  const int bx = blockIdx.x;
  const int b = bx >> 2;
  const int d = (bx & 3) * 256 + threadIdx.x;
  float Aa[D_STATE], h[D_STATE];
#pragma unroll
  for (int s = 0; s < D_STATE; ++s) {
    Aa[s] = -expf(A_log[d * D_STATE + s]);
    h[s] = 0.f;
  }
  for (int c = 0; c < NCH1; ++c) {             // serial fold over chunk summaries
    const size_t idx = (size_t)(b * NCH1 + c) * D_INNER + d;
    const float sdt = SDT[idx];
#pragma unroll
    for (int s = 0; s < D_STATE; ++s)
      h[s] = __expf(Aa[s] * sdt) * h[s] + HF[idx * D_STATE + s];
  }
  const float Dd = Dvec[d];
  const int l0 = NCH1 * CHUNK;                 // 1920
  for (int i = 0; i < CHUNK; ++i) {
    const int l = l0 + i;
    const size_t bl = (size_t)b * L + l;
    const int buf = i & 1;
    if (threadIdx.x < 32) sbc[buf][threadIdx.x] = XD[bl * 64 + DT_RANK + threadIdx.x];
    __syncthreads();
    const float dt = DT[bl * D_INNER + d];
    const float u  = UC[bl * D_INNER + d];
    float y = 0.f;
#pragma unroll
    for (int s = 0; s < D_STATE; ++s) {
      h[s] = __expf(dt * Aa[s]) * h[s] + dt * sbc[buf][s] * u;
      y += h[s] * sbc[buf][16 + s];
    }
    y += Dd * u;
    if (l >= L - PRED_LEN) {
      const float z = XZ[bl * (2 * D_INNER) + D_INNER + d];
      const float sz = z / (1.f + expf(-z));
      YL[((size_t)b * PRED_LEN + (l - (L - PRED_LEN))) * D_INNER + d] = y * sz;
    }
  }
}

// ---------------------------------------------------------------------------
// out = XO @ W_fc + b_fc  (768 x 7)
// ---------------------------------------------------------------------------
__global__ void k_head(const float* __restrict__ XO, const float* __restrict__ Wfc,
                       const float* __restrict__ bfc, float* __restrict__ out) {
  const int row = blockIdx.x;                  // b*96 + t
  const int c = threadIdx.x;
  if (c >= C_OUT) return;
  float acc = bfc[c];
  for (int dd = 0; dd < D_MODEL; ++dd)
    acc += XO[(size_t)row * D_MODEL + dd] * Wfc[dd * C_OUT + c];
  out[row * C_OUT + c] = acc;
}

// ---------------------------------------------------------------------------
extern "C" void kernel_launch(void* const* d_in, const int* in_sizes, int n_in,
                              void* d_out, int out_size, void* d_ws, size_t ws_size,
                              hipStream_t stream) {
  const float* x_enc    = (const float*)d_in[0];
  const float* x_mark   = (const float*)d_in[1];
  // d_in[2] (x_dec), d_in[3] (x_mark_dec) unused by the reference computation
  const float* W_in     = (const float*)d_in[4];
  const float* b_in     = (const float*)d_in[5];
  const float* W_inproj = (const float*)d_in[6];
  const float* conv_w   = (const float*)d_in[7];
  const float* conv_b   = (const float*)d_in[8];
  const float* W_xproj  = (const float*)d_in[9];
  const float* W_dt     = (const float*)d_in[10];
  const float* b_dt     = (const float*)d_in[11];
  const float* A_log    = (const float*)d_in[12];
  const float* Dv       = (const float*)d_in[13];
  const float* W_out    = (const float*)d_in[14];
  const float* W_fc     = (const float*)d_in[15];
  const float* b_fc     = (const float*)d_in[16];

  // Workspace layout (floats): ~320 MB total
  float* X   = (float*)d_ws;                         // BL x 512
  float* XZ  = X   + (size_t)BL * D_MODEL;           // BL x 2048
  float* UC  = XZ  + (size_t)BL * 2 * D_INNER;       // BL x 1024 (conv+silu)
  float* DT  = UC  + (size_t)BL * D_INNER;           // BL x 1024 (softplus dt)
  float* XD  = DT  + (size_t)BL * D_INNER;           // BL x 64   (dt_lo|B|C)
  float* YL  = XD  + (size_t)BL * 64;                // (8*96) x 1024
  float* XO  = YL  + (size_t)B_SZ * PRED_LEN * D_INNER;   // (8*96) x 512
  float* HF  = XO  + (size_t)B_SZ * PRED_LEN * D_MODEL;   // (8*15) x 1024 x 16
  float* SDT = HF  + (size_t)B_SZ * NCH1 * D_INNER * D_STATE; // (8*15) x 1024

  // 1) input transform
  k_in<<<dim3(BL, D_MODEL / 256), 256, 0, stream>>>(x_enc, x_mark, W_in, b_in, X);
  // 2) in_proj GEMM: (16384x512)*(512x2048), WMMA f16/f32-acc, B tile in LDS
  k_gemm<<<dim3(BL / 128, (2 * D_INNER) / 16), 256, 0, stream>>>(
      X, W_inproj, XZ, D_MODEL, 2 * D_INNER, 2 * D_INNER, D_MODEL);
  // 3) depthwise causal conv + SiLU
  k_conv<<<dim3(BL, D_INNER / 256), 256, 0, stream>>>(XZ, conv_w, conv_b, UC);
  // 4) x_proj (skinny N=64, VALU)
  k_xproj<<<dim3(BL), 64, 0, stream>>>(UC, W_xproj, XD);
  // 5) dt_proj + softplus (skinny K=32, VALU)
  k_dt<<<dim3(BL, D_INNER / 256), 256, 0, stream>>>(XD, W_dt, b_dt, DT);
  // 6a) chunked scan pass 1: chunks 0..14 in parallel (no y)
  k_scan1<<<dim3(B_SZ * 4, NCH1), 256, 0, stream>>>(UC, DT, XD, A_log, HF, SDT);
  // 6b) fold summaries + replay chunk 15, emit last-96 y*silu(z)
  k_scan2<<<dim3(B_SZ * 4), 256, 0, stream>>>(XZ, UC, DT, XD, A_log, Dv, HF, SDT, YL);
  // 7) out_proj GEMM on last-96 rows only: (768x1024)*(1024x512), WMMA
  k_gemm<<<dim3((B_SZ * PRED_LEN) / 128, D_MODEL / 16), 256, 0, stream>>>(
      YL, W_out, XO, D_INNER, D_MODEL, D_MODEL, D_INNER);
  // 8) head
  k_head<<<dim3(B_SZ * PRED_LEN), 32, 0, stream>>>(XO, W_fc, b_fc, (float*)d_out);
}
